// CodecVQAutoEncoder_28561532518543
// MI455X (gfx1250) — matrix operand order repaired and checked by reading, not compiled, not run on hardware
//
#include <hip/hip_runtime.h>

// ---------------- types / helpers ----------------
typedef __attribute__((ext_vector_type(16))) __bf16 v16bf;
typedef __attribute__((ext_vector_type(8)))  float  v8f;

union BFV { v16bf v; unsigned short u[16]; };

__device__ __forceinline__ unsigned short f2bf(float f) {
  union { float f; unsigned int u; } c; c.f = f;
  unsigned int r = c.u + 0x7FFFu + ((c.u >> 16) & 1u); // RNE
  return (unsigned short)(r >> 16);
}
__device__ __forceinline__ float lrelu(float v) { return v > 0.f ? v : 0.3f * v; }

__device__ __forceinline__ void async_wait0() {
#if __has_builtin(__builtin_amdgcn_s_wait_asynccnt)
  __builtin_amdgcn_s_wait_asynccnt(0);
#else
  asm volatile("s_wait_asynccnt 0" ::: "memory");
#endif
}

// ---------------- elementwise ----------------
__global__ void k_fill(float* p, float v, long n) {
  long i = (long)blockIdx.x * blockDim.x + threadIdx.x;
  if (i < n) p[i] = v;
}
__global__ void k_add(float* d, const float* __restrict__ s, long n) {
  long i = (long)blockIdx.x * blockDim.x + threadIdx.x;
  if (i < n) d[i] += s[i];
}
__global__ void k_cvt_flat(const float* __restrict__ in, unsigned short* __restrict__ out, long n) {
  long i = (long)blockIdx.x * blockDim.x + threadIdx.x;
  if (i < n) out[i] = f2bf(in[i]);
}

// ---------------- conv_pre: Cin=1, k=7, pad=3 (tiny K, direct) ----------------
__global__ void k_conv_pre(const float* __restrict__ x, const float* __restrict__ w,
                           const float* __restrict__ b, float* __restrict__ out, int T) {
  long i = (long)blockIdx.x * blockDim.x + threadIdx.x;
  long total = 4L * 32 * T;
  if (i >= total) return;
  int t = (int)(i % T); long r = i / T; int co = (int)(r % 32); int bb = (int)(r / 32);
  float s = b[co];
  const float* xb = x + (long)bb * T;
#pragma unroll
  for (int kk = 0; kk < 7; ++kk) {
    int ti = t + kk - 3;
    if (ti >= 0 && ti < T) s += w[co * 7 + kk] * xb[ti];
  }
  out[((long)bb * 32 + co) * T + t] = s;
}

// ---------------- weight convert: f32 [co][ci][k] -> bf16 [k][co][ci] ----------------
__global__ void k_wcvt(const float* __restrict__ w, unsigned short* __restrict__ o,
                       int Cout, int Cin, int K) {
  long i = (long)blockIdx.x * blockDim.x + threadIdx.x;
  long total = (long)Cout * Cin * K;
  if (i >= total) return;
  int kk = (int)(i % K); long r = i / K; int ci = (int)(r % Cin); int co = (int)(r / Cin);
  o[((long)kk * Cout + co) * Cin + ci] = f2bf(w[i]);
}

// ---------------- act prep: f32 [4][C][T] -> bf16 [4][T][C], fused prescale+lrelu ----
__global__ void k_prep(const float* __restrict__ in, unsigned short* __restrict__ out,
                       int C, int T, float prescale, int act) {
  __shared__ unsigned short tile[32][33];
  const int ct = C >> 5, tt = T >> 5;
  int bidx = blockIdx.x;
  int ctile = bidx % ct; int r = bidx / ct; int ttile = r % tt; int b = r / tt;
  const int tx = threadIdx.x & 31, ty = threadIdx.x >> 5;   // 32 x 8
#pragma unroll
  for (int i = 0; i < 32; i += 8) {
    int c = (ctile << 5) + ty + i;
    int t = (ttile << 5) + tx;
    float v = in[((long)b * C + c) * T + t] * prescale;
    if (act) v = lrelu(v);
    tile[ty + i][tx] = f2bf(v);          // tile[c_local][t_local]
  }
  __syncthreads();
#pragma unroll
  for (int i = 0; i < 32; i += 8) {
    int t = (ttile << 5) + ty + i;
    int c = (ctile << 5) + tx;
    out[((long)b * T + t) * C + c] = tile[tx][ty + i];
  }
}

// ---------------- implicit-GEMM conv1d via WMMA bf16 + async-LDS weight staging ----
// act: bf16 [4][Tin][Cin] (activated/prescaled).  w: bf16 [k][Cout][Cin].
// Workgroup: 8 waves share one (b, 32-co block); each wave owns a 32t x 32co tile.
// Per (tap, ci-chunk): waves 0..3 async-stage the 32co x 32ci weight tile (2KB) to LDS;
// every wave then does 4 global A b128 loads + 4 LDS B b128 loads + 4 WMMAs.
__global__ void k_conv_wmma(const unsigned short* __restrict__ act, float* out,
                            const unsigned short* __restrict__ w,
                            const float* __restrict__ bias,
                            const float* res,
                            int Cin, int Cout, int Tin, int Tout,
                            int K, int dil, int stride, int pad) {
  __shared__ unsigned short wtile[1024];    // [co 32][ci 32] bf16 = 2KB
  const int lane = threadIdx.x & 31;
  const int wv   = threadIdx.x >> 5;        // 0..7
  const int coT = Cout >> 5, tS = Tout >> 8;
  int bid = blockIdx.x;
  int co_t = bid % coT; int r1 = bid / coT;
  int t_s  = r1 % tS;   int b  = r1 / tS;
  const int co0 = co_t << 5;
  const int t0  = (t_s << 8) + (wv << 5);   // this wave's 32-t base
  const int hl = lane >> 4;                 // half-wave
  const int mr = lane & 15;

  v8f acc00 = {}, acc01 = {}, acc10 = {}, acc11 = {};
  const unsigned short* ab = act + (long)b * Tin * Cin;
  const int tvA = (t0 + mr) * stride - pad;
  const int tvB = (t0 + 16 + mr) * stride - pad;
  const uint4 z4 = make_uint4(0u, 0u, 0u, 0u);

  // async staging addressing: waves 0..3 move 512B each (16B per lane)
  const int lin  = (wv << 9) + (lane << 4); // byte offset within 2KB tile
  const int wrow = lin >> 6;                // co row (64B per row)
  const int wcol = lin & 63;                // byte within row
  const unsigned ldsa = (unsigned)(unsigned long long)(&wtile[0]) + (unsigned)lin;

  for (int kk = 0; kk < K; ++kk) {
    const int ta = tvA + kk * dil;
    const int tb = tvB + kk * dil;
    const bool okA = (ta >= 0) && (ta < Tin);
    const bool okB = (tb >= 0) && (tb < Tin);
    const unsigned short* arA = ab + (long)ta * Cin;
    const unsigned short* arB = ab + (long)tb * Cin;
    for (int cb = 0; cb < Cin; cb += 32) {
      if (wv < 4) {                         // wave-uniform branch
        const unsigned short* gp = w + ((long)kk * Cout + co0 + wrow) * Cin + cb;
        unsigned long long ga = (unsigned long long)gp + (unsigned long long)wcol;
        asm volatile("global_load_async_to_lds_b128 %0, %1, off"
                     :: "v"(ldsa), "v"(ga) : "memory");
        async_wait0();
      }
      __syncthreads();
      // A fragments (both t tiles): K slots 0..7 <- hl*8+j, 8..15 <- 16+hl*8+j
      BFV a0, a1;
      if (okA) {
        const uint4* ap = (const uint4*)(arA + cb + hl * 8);
        *(uint4*)&a0.u[0] = ap[0];
        *(uint4*)&a0.u[8] = ap[2];
      } else { *(uint4*)&a0.u[0] = z4; *(uint4*)&a0.u[8] = z4; }
      if (okB) {
        const uint4* ap = (const uint4*)(arB + cb + hl * 8);
        *(uint4*)&a1.u[0] = ap[0];
        *(uint4*)&a1.u[8] = ap[2];
      } else { *(uint4*)&a1.u[0] = z4; *(uint4*)&a1.u[8] = z4; }
      // B fragments from LDS: col n = mr (+16), K slots 0..7 <- hl*8+j, 8..15 <- 16+hl*8+j
      BFV b0, b1;
      const unsigned short* l0 = &wtile[mr * 32 + hl * 8];
      const unsigned short* l1 = &wtile[(16 + mr) * 32 + hl * 8];
      *(uint4*)&b0.u[0] = ((const uint4*)l0)[0];
      *(uint4*)&b0.u[8] = ((const uint4*)l0)[2];
      *(uint4*)&b1.u[0] = ((const uint4*)l1)[0];
      *(uint4*)&b1.u[8] = ((const uint4*)l1)[2];
      acc00 = __builtin_amdgcn_wmma_f32_16x16x32_bf16(false, a0.v, false, b0.v, (short)0, acc00, false, false);
      acc01 = __builtin_amdgcn_wmma_f32_16x16x32_bf16(false, a0.v, false, b1.v, (short)0, acc01, false, false);
      acc10 = __builtin_amdgcn_wmma_f32_16x16x32_bf16(false, a1.v, false, b0.v, (short)0, acc10, false, false);
      acc11 = __builtin_amdgcn_wmma_f32_16x16x32_bf16(false, a1.v, false, b1.v, (short)0, acc11, false, false);
      __syncthreads();                      // protect wtile before next staging
    }
  }
  // D layout: slot i -> t = tbase + i + 8*hl; col = co0 (+16) + mr
  const int coA = co0 + mr, coB = co0 + 16 + mr;
  const float bvA = bias[coA], bvB = bias[coB];
  float* oA = out + ((long)b * Cout + coA) * Tout;
  float* oB = out + ((long)b * Cout + coB) * Tout;
  const float* rA = res ? res + ((long)b * Cout + coA) * Tout : (const float*)0;
  const float* rB = res ? res + ((long)b * Cout + coB) * Tout : (const float*)0;
#pragma unroll
  for (int i = 0; i < 8; ++i) {
    int t1 = t0 + i + 8 * hl;
    int t2 = t1 + 16;
    float v00 = acc00[i] + bvA, v01 = acc01[i] + bvB;
    float v10 = acc10[i] + bvA, v11 = acc11[i] + bvB;
    if (rA) { v00 += rA[t1]; v10 += rA[t2]; v01 += rB[t1]; v11 += rB[t2]; }
    oA[t1] = v00; oA[t2] = v10;
    oB[t1] = v01; oB[t2] = v11;
  }
}

// ---------------- VQ ----------------
// yk/yout layout: [n=1024][c=128], n = b*256 + t
__global__ void k_vq_init(const float* __restrict__ z, float* __restrict__ yk,
                          float* __restrict__ yout, float* loss_acc, float* perp_acc) {
  int gid = blockIdx.x * 256 + threadIdx.x;      // 131072
  int t = gid & 255; int r = gid >> 8; int c = r & 127; int b = r >> 7;
  int n = (b << 8) + t;
  float v = z[gid];
  yk[n * 128 + c] = v;
  yout[n * 128 + c] = 0.f;
  if (gid == 0) { *loss_acc = 0.f; *perp_acc = 0.f; }
}

__global__ void k_cb_norms(const float* __restrict__ E, float* __restrict__ nrm) {
  int row = blockIdx.x * 256 + threadIdx.x;      // 16*1024 rows
  const float* e = E + (long)row * 128;
  float s = 0.f;
  for (int c = 0; c < 128; ++c) s += e[c] * e[c];
  nrm[row] = s;
}

__global__ void k_zero_counts(int* counts) {
  int i = blockIdx.x * 256 + threadIdx.x;
  if (i < 1024) counts[i] = 0;
}

// score(n,e) = |E_e|^2 - 2 * <yk_n, E_e>  (|z|^2 const per row -> dropped for argmin)
__global__ void k_vq_argmin(const unsigned short* __restrict__ ykb,
                            const unsigned short* __restrict__ Eb,
                            const float* __restrict__ en2, int* __restrict__ oidx) {
  __shared__ float smin[16][128];
  __shared__ int   sidx[16][128];
  const int lane = threadIdx.x & 31;
  const int w = threadIdx.x >> 5;          // 8 waves
  const int hl = lane >> 4;
  const int mr = lane & 15;
  const int n0 = blockIdx.x << 4;          // 64 blocks x 16 tokens

  BFV afr[4];                              // cached token tile (K = 128)
  const unsigned short* zr = ykb + (long)(n0 + mr) * 128;
#pragma unroll
  for (int q = 0; q < 4; ++q) {
    const uint4* ap = (const uint4*)(zr + q * 32 + hl * 8);
    *(uint4*)&afr[q].u[0] = ap[0];
    *(uint4*)&afr[q].u[8] = ap[2];
  }
  float best[8]; int bid[8];
#pragma unroll
  for (int i = 0; i < 8; ++i) { best[i] = 3.4e38f; bid[i] = 0; }

  for (int et = w; et < 64; et += 8) {
    const int e0 = et << 4;
    v8f acc = {};
#pragma unroll
    for (int q = 0; q < 4; ++q) {
      BFV bf;
      const unsigned short* ep = Eb + (long)(e0 + mr) * 128 + q * 32 + hl * 8;
      *(uint4*)&bf.u[0] = ((const uint4*)ep)[0];
      *(uint4*)&bf.u[8] = ((const uint4*)ep)[2];
      acc = __builtin_amdgcn_wmma_f32_16x16x32_bf16(false, afr[q].v, false, bf.v,
                                                    (short)0, acc, false, false);
    }
    const int e = e0 + mr;
    const float n2 = en2[e];
#pragma unroll
    for (int i = 0; i < 8; ++i) {
      float d = n2 - 2.0f * acc[i];
      if (d < best[i]) { best[i] = d; bid[i] = e; }
    }
  }
#pragma unroll
  for (int i = 0; i < 8; ++i) {
    int m = i + 8 * hl;                    // D row
    smin[m][(w << 4) + mr] = best[i];
    sidx[m][(w << 4) + mr] = bid[i];
  }
  __syncthreads();
  const int m = threadIdx.x >> 4;
  const int seg = threadIdx.x & 15;
  float bv = 3.5e38f; int bi = 0;
#pragma unroll
  for (int j = 0; j < 8; ++j) {
    float v = smin[m][seg * 8 + j];
    int  id = sidx[m][seg * 8 + j];
    if (v < bv) { bv = v; bi = id; }
  }
  __syncthreads();
  smin[m][seg] = bv; sidx[m][seg] = bi;
  __syncthreads();
  if (seg == 0) {
    float vv = smin[m][0]; int ii = sidx[m][0];
    for (int j = 1; j < 16; ++j)
      if (smin[m][j] < vv) { vv = smin[m][j]; ii = sidx[m][j]; }
    oidx[n0 + m] = ii;
  }
}

__global__ void k_vq_update(const float* __restrict__ E, const int* __restrict__ idx,
                            float* __restrict__ yk, float* __restrict__ yout,
                            float* __restrict__ partial, int* __restrict__ counts) {
  __shared__ float red[256];
  int gid = blockIdx.x * 256 + threadIdx.x;      // 512 blocks
  int c = gid & 127, n = gid >> 7;
  int e = idx[n];
  float q = E[(long)e * 128 + c];
  float r = yk[gid];
  float d = q - r;                                // zq - zt (pre-update residual)
  yout[gid] += q;
  yk[gid] = r - q;
  if (c == 0) atomicAdd(counts + e, 1);
  red[threadIdx.x] = d * d;
  __syncthreads();
  for (int s = 128; s > 0; s >>= 1) {
    if (threadIdx.x < s) red[threadIdx.x] += red[threadIdx.x + s];
    __syncthreads();
  }
  if (threadIdx.x == 0) partial[blockIdx.x] = red[0];   // deterministic
}

__global__ void k_vq_stats(const int* __restrict__ counts, const float* __restrict__ partial,
                           float* loss_acc, float* perp_acc) {
  __shared__ float ra[256], rb[256];
  int tid = threadIdx.x;
  float h = 0.f;
  for (int e = tid; e < 1024; e += 256) {
    float p = (float)counts[e] * (1.0f / 1024.0f);
    h += p * logf(p + 1e-10f);
  }
  ra[tid] = h;
  rb[tid] = partial[tid] + partial[tid + 256];
  __syncthreads();
  for (int s = 128; s > 0; s >>= 1) {
    if (tid < s) { ra[tid] += ra[tid + s]; rb[tid] += rb[tid + s]; }
    __syncthreads();
  }
  if (tid == 0) {
    float perp = expf(-ra[0]);
    float loss = 1.25f * rb[0] * (1.0f / (1024.0f * 128.0f)); // (1+beta)*mean(diff^2)
    *loss_acc += loss * (1.0f / 16.0f);
    *perp_acc += perp * (1.0f / 16.0f);
  }
}

__global__ void k_vq_final(const float* __restrict__ yout, const float* loss_acc,
                           const float* perp_acc, float* __restrict__ out) {
  long gid = (long)blockIdx.x * 256 + threadIdx.x;
  if (gid < 131072) {
    int t = (int)(gid & 255); int r = (int)(gid >> 8); int c = r & 127; int b = r >> 7;
    int n = (b << 8) + t;
    out[gid] = yout[n * 128 + c];
  } else if (gid == 131072) {
    out[131072] = *loss_acc;
  } else if (gid == 131073) {
    out[131073] = *perp_acc;
  }
}

// ---------------- host orchestration ----------------
extern "C" void kernel_launch(void* const* d_in, const int* in_sizes, int n_in,
                              void* d_out, int out_size, void* d_ws, size_t ws_size,
                              hipStream_t stream) {
  (void)in_sizes; (void)n_in; (void)out_size; (void)ws_size;
  static const int CH[5]   = {32, 64, 128, 256, 512};
  static const int TSZ[5]  = {65536, 32768, 16384, 2048, 256};
  static const int RATE[4] = {2, 2, 8, 8};
  static const int DK[4]   = {3, 3, 15, 15};
  static const int RBK[3]  = {3, 7, 11};
  static const int DILS[3] = {1, 3, 5};

  // ---- decode inputs (recursive dict insertion order) ----
  int p = 0;
  const float* x      = (const float*)d_in[p++];
  const float* cpre_w = (const float*)d_in[p++];
  const float* cpre_b = (const float*)d_in[p++];
  const float *dw[4], *db[4];
  for (int i = 0; i < 4; ++i) { dw[i] = (const float*)d_in[p++]; db[i] = (const float*)d_in[p++]; }
  const float *rbw[12][6], *rbb[12][6];   // per rb: c1[0..2]{w,b}, c2[0..2]{w,b}
  for (int r = 0; r < 12; ++r)
    for (int j = 0; j < 6; ++j) { rbw[r][j] = (const float*)d_in[p++]; rbb[r][j] = (const float*)d_in[p++]; }
  const float* cpost_w = (const float*)d_in[p++];
  const float* cpost_b = (const float*)d_in[p++];
  const float* cbs     = (const float*)d_in[p++];

  // ---- workspace layout ----
  const long M = 8388608L;                      // floats: 4*32*65536
  float* ws   = (float*)d_ws;
  float* bufA = ws;                             // stage input / ping
  float* bufS = ws + M;                         // resblock-sum accumulator, later z
  float* bufL = ws + 2 * M;                     // x_local (resblock running state)
  float* bufT = ws + 3 * M;                     // xt temp / pong
  unsigned short* wb   = (unsigned short*)(ws + 4 * M);   // bf16 weights
  const long WCAP = 16777216L;
  unsigned short* actb = wb + WCAP;             // bf16 activations [4][T][C]
  unsigned short* cbb  = actb + 8388608L;       // bf16 codebooks
  unsigned short* ykb  = cbb + 2097152L;        // bf16 residual tokens
  float* vqf   = (float*)(ykb + 131072L + 64);
  float* yk    = vqf;                           // [1024][128]
  float* yout  = yk + 131072;
  float* nrm   = yout + 131072;                 // 16*1024
  float* part  = nrm + 16384;                   // 512
  float* lossA = part + 512;
  float* perpA = lossA + 1;
  int*   counts= (int*)(perpA + 1);             // 1024
  int*   idxb  = counts + 1024;                 // 1024

  // ---- convert conv weights to bf16 [k][co][ci] ----
  size_t woff = 0;
  auto alloc_w = [&](long n) -> size_t { size_t o = woff; woff += (size_t)((n + 63L) & ~63L); return o; };
  auto cvt = [&](const float* w, size_t off, int Cout, int Cin, int K) {
    long total = (long)Cout * Cin * K;
    k_wcvt<<<(int)((total + 255) / 256), 256, 0, stream>>>(w, wb + off, Cout, Cin, K);
  };
  size_t off_down[4], off_rb[12][6], off_post;
  for (int i = 0; i < 4; ++i) {
    off_down[i] = alloc_w((long)CH[i + 1] * CH[i] * DK[i]);
    cvt(dw[i], off_down[i], CH[i + 1], CH[i], DK[i]);
  }
  for (int i = 0; i < 4; ++i)
    for (int j = 0; j < 3; ++j) {
      int r = i * 3 + j, C = CH[i], K = RBK[j];
      for (int q = 0; q < 6; ++q) {
        off_rb[r][q] = alloc_w((long)C * C * K);
        cvt(rbw[r][q], off_rb[r][q], C, C, K);
      }
    }
  off_post = alloc_w(128L * 512 * 7);
  cvt(cpost_w, off_post, 128, 512, 7);

  auto ew = [](long n) { return (int)((n + 255) / 256); };
  auto prep = [&](const float* in, int C, int T, float prescale, int act) {
    int blocks = (C >> 5) * (T >> 5) * 4;
    k_prep<<<blocks, 256, 0, stream>>>(in, actb, C, T, prescale, act);
  };
  auto conv = [&](float* out, size_t wo, const float* bias, const float* res,
                  int Cin, int Cout, int Tin, int Tout, int K, int dil, int stride, int pad) {
    int blocks = 4 * (Cout / 32) * (Tout / 256);    // exact grid, 8 waves/block
    k_conv_wmma<<<blocks, 256, 0, stream>>>(actb, out, wb + wo, bias, res,
                                            Cin, Cout, Tin, Tout, K, dil, stride, pad);
  };

  // ---- encoder ----
  k_conv_pre<<<ew(4L * 32 * 65536), 256, 0, stream>>>(x, cpre_w, cpre_b, bufA, 65536);

  float* X = bufA; float* TMP = bufT;
  for (int i = 0; i < 4; ++i) {
    const int C = CH[i], T = TSZ[i];
    const long n = 4L * C * T;
    k_fill<<<ew(n), 256, 0, stream>>>(bufS, 0.f, n);
    for (int j = 0; j < 3; ++j) {
      const int K = RBK[j], r = i * 3 + j;
      for (int dd = 0; dd < 3; ++dd) {
        const int dil = DILS[dd];
        const float* xin = (dd == 0) ? X : bufL;     // x_local state
        prep(xin, C, T, 1.0f, 1);                    // lrelu(x_local)
        conv(TMP, off_rb[r][dd], rbb[r][dd], nullptr,
             C, C, T, T, K, dil, 1, (K * dil - dil) / 2);
        prep(TMP, C, T, 1.0f, 1);                    // lrelu(xt)
        conv(bufL, off_rb[r][3 + dd], rbb[r][3 + dd], xin,
             C, C, T, T, K, 1, 1, (K - 1) / 2);
      }
      k_add<<<ew(n), 256, 0, stream>>>(bufS, bufL, n);
    }
    prep(bufS, C, T, 1.0f / 3.0f, 1);                // lrelu(xs/3)
    conv(TMP, off_down[i], db[i], nullptr,
         C, CH[i + 1], T, TSZ[i + 1], DK[i], 1, RATE[i], DK[i] / 2);
    float* sw = X; X = TMP; TMP = sw;
  }
  prep(X, 512, 256, 1.0f, 1);                        // lrelu(x)
  conv(bufS, off_post, cpost_b, nullptr, 512, 128, 256, 256, 7, 1, 1, 3);

  // ---- multistage VQ ----
  k_vq_init<<<512, 256, 0, stream>>>(bufS, yk, yout, lossA, perpA);
  k_cvt_flat<<<ew(2097152L), 256, 0, stream>>>(cbs, cbb, 2097152L);
  k_cb_norms<<<64, 256, 0, stream>>>(cbs, nrm);
  for (int st = 0; st < 16; ++st) {
    k_zero_counts<<<4, 256, 0, stream>>>(counts);
    k_cvt_flat<<<512, 256, 0, stream>>>(yk, ykb, 131072L);
    const float* E = cbs + (long)st * 131072;
    k_vq_argmin<<<64, 256, 0, stream>>>(ykb, cbb + (long)st * 131072, nrm + st * 1024, idxb);
    k_vq_update<<<512, 256, 0, stream>>>(E, idxb, yk, yout, part, counts);
    k_vq_stats<<<1, 256, 0, stream>>>(counts, part, lossA, perpA);
  }
  k_vq_final<<<513, 256, 0, stream>>>(yout, lossA, perpA, (float*)d_out);
}
